// InitDispNet_39651138077351
// MI455X (gfx1250) — matrix operand order repaired and checked by reading, not compiled, not run on hardware
//
#include <hip/hip_runtime.h>

typedef __attribute__((ext_vector_type(16))) _Float16 v16h;
typedef __attribute__((ext_vector_type(8)))  float    v8f;
typedef __attribute__((ext_vector_type(4)))  unsigned int v4u;
typedef __attribute__((ext_vector_type(8)))  int    v8i;
typedef __attribute__((ext_vector_type(4)))  int    v4i;

#if defined(__has_builtin)
#if __has_builtin(__builtin_amdgcn_tensor_load_to_lds)
#define HAVE_TDM 1
#endif
#endif

__device__ __forceinline__ float leaky(float v) { return v > 0.f ? v : 0.2f * v; }

#ifdef HAVE_TDM
// Issue a TDM 2D tile load: rows x width elements of f32, row stride rowStride
// (elements) in global memory, packed contiguously at LDS byte offset ldsOff.
// D# layout per CDNA5 ISA ch.8 (group0: count/lds/global/type, group1: dims).
__device__ __forceinline__ void tdm_load_2d_f32(unsigned int ldsOff,
                                                const float* gptr,
                                                unsigned int width,
                                                unsigned int rows,
                                                unsigned int rowStride) {
  unsigned long long ga = (unsigned long long)(uintptr_t)gptr;
  v4u g0;
  g0[0] = 1u;                                   // count=1 (valid), user mode
  g0[1] = ldsOff;                               // lds_addr [63:32]
  g0[2] = (unsigned int)ga;                     // global_addr low
  g0[3] = (unsigned int)((ga >> 32) & 0x1FFFFFFu) | (2u << 30); // hi + type=2
  v8i g1;
  g1[0] = (int)(2u << 16);                      // data_size=2 (4B), no mask
  g1[1] = (int)((width & 0xFFFFu) << 16);       // tensor_dim0 [15:0]
  g1[2] = (int)((width >> 16) | ((rows & 0xFFFFu) << 16)); // dim0 hi, dim1 lo
  g1[3] = (int)((rows >> 16) | (width << 16));  // dim1 hi, tile_dim0
  g1[4] = (int)rows;                            // tile_dim1 (tile_dim2=0)
  g1[5] = (int)rowStride;                       // tensor_dim0_stride low32
  g1[6] = 0;                                    // stride hi, dim1_stride lo
  g1[7] = 0;
  v4i gz = {0, 0, 0, 0};
#if __clang_major__ >= 23
  v8i gz8 = {0, 0, 0, 0, 0, 0, 0, 0};
  __builtin_amdgcn_tensor_load_to_lds(g0, g1, gz, gz, gz8, 0);
#else
  __builtin_amdgcn_tensor_load_to_lds(g0, g1, gz, gz, 0);
#endif
}
#endif

// ---------------------------------------------------------------------------
// Branch conv: 4x4 conv (cin=16 -> cout=24), stride (4, SX), pad-left PADL in W,
// + bias + leaky, then 1x1 conv (24->24) + bias + leaky.
// im2col GEMM on WMMA f16: M=24 (pad 32), N=16 pixels per wave, K=256.
// LDS tiles stored transposed ([N][K], 16B-aligned padded rows) so every
// WMMA fragment is contiguous ds_load_b128 traffic.
// ---------------------------------------------------------------------------
template <int SX, int PADL, int WO>
__global__ __launch_bounds__(128) void conv_branch_kernel(
    const float* __restrict__ in,    // [B,16,384,1280]
    const float* __restrict__ w_em,  // [24,16,4,4] = [24,256]
    const float* __restrict__ b_em,  // [24]
    const float* __restrict__ w_rc,  // [24,24]
    const float* __restrict__ b_rc,  // [24]
    float* __restrict__ outp)        // [B,24,96,WO]
{
  constexpr int CIN = 16, COUT = 24, H = 384, W = 1280, HO = 96;
  constexpr int NTX = WO / 16;
  constexpr int PP = 264;  // patch row pitch (halves): 256 + 8 pad, 16B-mult
  constexpr int EP = 40;   // em row pitch (halves): 32 + 8 pad, 16B-mult

  __shared__ _Float16 lds_w[32 * 256];       // em weights, M-padded to 32
  __shared__ _Float16 lds_wrc[32 * 32];      // rc weights, padded to 32x32
  __shared__ _Float16 lds_patch[4][16 * PP]; // per-wave im2col tile [N][K]
  __shared__ _Float16 lds_em[4][16 * EP];    // per-wave intermediate [N][mi]

  const int tid  = threadIdx.x;
  const int lane = tid & 31;
  const int wv   = tid >> 5;
  const int lh   = lane & 15;   // N column / M row within half
  const int hi   = lane >> 4;   // half selector

  // ---- stage weights (cooperative, whole block) ----
  for (int i = tid; i < 32 * 256; i += 128) {
    int m = i >> 8, k = i & 255;
    lds_w[i] = (m < COUT) ? (_Float16)w_em[m * 256 + k] : (_Float16)0.f;
  }
  for (int i = tid; i < 32 * 32; i += 128) {
    int mo = i >> 5, mi = i & 31;
    lds_wrc[i] = (mo < COUT && mi < COUT) ? (_Float16)w_rc[mo * 24 + mi]
                                          : (_Float16)0.f;
  }

  // ---- this wave's output tile ----
  const int tile = blockIdx.x * 4 + wv;
  const int tx = tile % NTX;
  const int t2 = tile / NTX;
  const int h  = t2 % HO;
  const int b  = t2 / HO;
  const int x0 = tx * 16;

  // ---- stage im2col patch transposed: patch[n][k], k = c*16 + ky*4 + kx ----
  _Float16* patch = lds_patch[wv];
  for (int idx = lane; idx < 256 * 16; idx += 32) {
    int k = idx >> 4, n = idx & 15;
    int c = k >> 4, ky = (k >> 2) & 3, kx = k & 3;
    int col = (x0 + n) * SX + kx - PADL;
    int row = h * 4 + ky;
    float v = 0.f;
    if (col >= 0 && col < W)
      v = in[(((long)b * CIN + c) * H + row) * W + col];
    patch[n * PP + k] = (_Float16)v;
  }
  __syncthreads();

  // ---- em conv GEMM: K=256 in 8 steps of K=32 ----
  const _Float16* prow = patch + lh * PP + hi * 16; // this lane's B row base
  v8f c0 = {}; v8f c1 = {};
#pragma unroll
  for (int s = 0; s < 8; ++s) {
    v16h bf, a0, a1;
#pragma unroll
    for (int e = 0; e < 16; ++e) {
      // B layout: lanes 0-15 hold K=base+0..15, lanes 16-31 K=base+16..31
      bf[e] = prow[s * 32 + e];
      // A layout: lanes 0-15 hold K=base+0..15, lanes 16-31 K=base+16..31
      int ka = s * 32 + hi * 16 + e;
      a0[e] = lds_w[(lh) * 256 + ka];
      a1[e] = lds_w[(16 + lh) * 256 + ka];
    }
    c0 = __builtin_amdgcn_wmma_f32_16x16x32_f16(false, a0, false, bf,
                                                (short)0, c0, false, false);
    c1 = __builtin_amdgcn_wmma_f32_16x16x32_f16(false, a1, false, bf,
                                                (short)0, c1, false, false);
  }

  // ---- bias + leaky, restage transposed [n][mi(pad 32)] for rc GEMM ----
  _Float16* em = lds_em[wv] + lh * EP;
#pragma unroll
  for (int i = 0; i < 8; ++i) {
    int m0 = i + hi * 8;                       // 0..15
    em[m0] = (_Float16)leaky((float)c0[i] + b_em[m0]);
    int m1 = 16 + i + hi * 8;                  // 16..31
    float e1 = (m1 < COUT) ? leaky((float)c1[i] + b_em[m1]) : 0.f;
    em[m1] = (_Float16)e1;
  }
  __syncthreads();

  // ---- rc 1x1 conv: 24x24 GEMM, single K=32 step ----
  v16h brc, ar0, ar1;
#pragma unroll
  for (int e = 0; e < 16; ++e) {
    brc[e] = em[hi * 16 + e];
    int ka = hi * 16 + e;
    ar0[e] = lds_wrc[(lh) * 32 + ka];
    ar1[e] = lds_wrc[(16 + lh) * 32 + ka];
  }
  v8f z = {};
  v8f d0 = __builtin_amdgcn_wmma_f32_16x16x32_f16(false, ar0, false, brc,
                                                  (short)0, z, false, false);
  v8f d1 = __builtin_amdgcn_wmma_f32_16x16x32_f16(false, ar1, false, brc,
                                                  (short)0, z, false, false);

  // ---- bias + leaky + store fl/fr tile ----
  const int x = x0 + lh;
#pragma unroll
  for (int i = 0; i < 8; ++i) {
    int m0 = i + hi * 8;
    outp[(((long)b * COUT + m0) * HO + h) * WO + x] =
        leaky((float)d0[i] + b_rc[m0]);
    int m1 = 16 + i + hi * 8;
    if (m1 < COUT)
      outp[(((long)b * COUT + m1) * HO + h) * WO + x] =
          leaky((float)d1[i] + b_rc[m1]);
  }
}

// ---------------------------------------------------------------------------
// Fused cost volume + min/argmin + 1x1 head + output assembly.
// One workgroup per (b,h) row; 320 threads = one output pixel each.
// fr row (24x1280 f32) + fl row (24x320) staged in LDS by the Tensor Data
// Mover (one 2D-tile DMA each, issued by wave 0, TENSORcnt-tracked);
// ~155 KB total, within CDNA5's 320 KB per-WGP LDS.
// ---------------------------------------------------------------------------
__global__ __launch_bounds__(320) void cost_head_kernel(
    const float* __restrict__ fl,    // [B,24,96,320]
    const float* __restrict__ fr,    // [B,24,96,1280]
    const float* __restrict__ w_tf,  // [13,25]
    const float* __restrict__ b_tf,  // [13]
    float* __restrict__ out,         // [B,16,96,320]
    float* __restrict__ cost)        // [B,64,96,320]
{
  constexpr int C = 24, HO = 96, WQ = 320, WR = 1280, D = 64;
  extern __shared__ float lds[];
  float* s_fr = lds;              // [24][1280]
  float* s_fl = s_fr + C * WR;    // [24][320]
  float* s_wt = s_fl + C * WQ;    // [13*25] weights then [13] bias

  const int tid = threadIdx.x;
  const int h = blockIdx.x % HO;
  const int b = blockIdx.x / HO;

  const float* frp = fr + (((long)b * C) * HO + h) * WR;
  const float* flp = fl + (((long)b * C) * HO + h) * WQ;

#ifdef HAVE_TDM
  if (tid < 32) {   // one wave issues the DMAs (TDM ignores EXEC; avoid dups)
    tdm_load_2d_f32((unsigned int)(uintptr_t)s_fr, frp, WR, C,
                    (unsigned int)(HO * WR));
    tdm_load_2d_f32((unsigned int)(uintptr_t)s_fl, flp, WQ, C,
                    (unsigned int)(HO * WQ));
    __builtin_amdgcn_s_wait_tensorcnt(0);
  }
#else
  for (int i = tid; i < C * WR; i += 320) {
    int c = i / WR, xx = i - c * WR;
    s_fr[i] = frp[(long)c * HO * WR + xx];
  }
  for (int i = tid; i < C * WQ; i += 320) {
    int c = i / WQ, xx = i - c * WQ;
    s_fl[i] = flp[(long)c * HO * WQ + xx];
  }
#endif
  for (int i = tid; i < 13 * 25 + 13; i += 320)
    s_wt[i] = (i < 13 * 25) ? w_tf[i] : b_tf[i - 13 * 25];
  __syncthreads();

  const int x = tid;              // 0..319
  float flv[C];
#pragma unroll
  for (int c = 0; c < C; ++c) flv[c] = s_fl[c * WQ + x];

  float best = 3.4e38f;
  int bestd = 0;
  float* costp = cost + (((long)b * D) * HO + h) * WQ + x;
  for (int d = 0; d < D; ++d) {
    int idx = 4 * x - d + 1;
    idx = idx < 0 ? 0 : (idx > WR - 1 ? WR - 1 : idx);
    float s = 0.f;
#pragma unroll
    for (int c = 0; c < C; ++c) s += fabsf(flv[c] - s_fr[c * WR + idx]);
    costp[(long)d * HO * WQ] = s;
    if (s < best) { best = s; bestd = d; }   // strict < keeps first (argmin)
  }

  // head: p = [cost_f, fl], 25 -> 13 with leaky
  float p[25];
  p[0] = best;
#pragma unroll
  for (int c = 0; c < C; ++c) p[1 + c] = flv[c];

  float* outp = out + (((long)b * 16) * HO + h) * WQ + x;
  outp[0] = (float)bestd;                  // d_init
  outp[(long)1 * HO * WQ] = 0.f;
  outp[(long)2 * HO * WQ] = 0.f;
#pragma unroll
  for (int j = 0; j < 13; ++j) {
    float acc = s_wt[13 * 25 + j];
#pragma unroll
    for (int i = 0; i < 25; ++i) acc += s_wt[j * 25 + i] * p[i];
    outp[(long)(3 + j) * HO * WQ] = leaky(acc);
  }
}

// ---------------------------------------------------------------------------
extern "C" void kernel_launch(void* const* d_in, const int* in_sizes, int n_in,
                              void* d_out, int out_size, void* d_ws, size_t ws_size,
                              hipStream_t stream) {
  (void)in_sizes; (void)n_in; (void)out_size; (void)ws_size;
  const float* fL   = (const float*)d_in[0];
  const float* fR   = (const float*)d_in[1];
  // d_in[2] = max_disp (int scalar, == 64 in this problem; hardcoded as D=64)
  const float* w_em = (const float*)d_in[3];
  const float* b_em = (const float*)d_in[4];
  const float* w_rc = (const float*)d_in[5];
  const float* b_rc = (const float*)d_in[6];
  const float* w_tf = (const float*)d_in[7];
  const float* b_tf = (const float*)d_in[8];

  float* fl = (float*)d_ws;                     // [2,24,96,320]
  float* fr = fl + (long)2 * 24 * 96 * 320;     // [2,24,96,1280]

  float* out  = (float*)d_out;                  // [2,16,96,320] = 983040
  float* cost = out + (long)2 * 16 * 96 * 320;  // [2,64,96,320]

  // left branch: stride (4,4), VALID, Wout=320 -> 2*96*20 = 3840 tiles / 4
  conv_branch_kernel<4, 0, 320><<<960, 128, 0, stream>>>(
      fL, w_em, b_em, w_rc, b_rc, fl);
  // right branch: stride (4,1), SAME pad (pl=1), Wout=1280 -> 15360 tiles / 4
  conv_branch_kernel<1, 1, 1280><<<3840, 128, 0, stream>>>(
      fR, w_em, b_em, w_rc, b_rc, fr);

  size_t shmem = (size_t)(24 * 1280 + 24 * 320 + 13 * 25 + 13) * sizeof(float);
  cost_head_kernel<<<192, 320, shmem, stream>>>(fl, fr, w_tf, b_tf, out, cost);
}
